// BERTResidualGATv2ContextGatedFusion_17540646437510
// MI455X (gfx1250) — matrix operand order repaired and checked by reading, not compiled, not run on hardware
//
#include <hip/hip_runtime.h>
#include <math.h>

// ---------------- problem constants (match reference) ----------------
constexpr int BATCH = 64, SEQ = 512, DBERT = 768, HID = 256, NHEADS = 4;
constexpr int NW = 24576, NE = 49152, NLAB = 2;
constexpr int NTOK = BATCH * SEQ;      // 32768
constexpr int ETOT = NE + NW;          // 73728 (edges + self loops)
constexpr float LN_EPS = 1e-5f;
constexpr float NEG_SLOPE = 0.2f;

// ---------------- CDNA5 WMMA types ----------------
typedef __bf16 bf16;
typedef __attribute__((ext_vector_type(16))) __bf16 v16bf;
typedef __attribute__((ext_vector_type(8)))  float  v8f;
typedef __attribute__((ext_vector_type(4)))  float  f4;

union FragAB { v16bf v; unsigned int u[8]; };

__device__ __forceinline__ float waveReduceSum(float v) {
#pragma unroll
  for (int off = 16; off > 0; off >>= 1) v += __shfl_xor(v, off, 32);
  return v;
}

__device__ __forceinline__ void atomicMaxFloat(float* addr, float val) {
  if (val >= 0.0f) atomicMax((int*)addr, __float_as_int(val));
  else             atomicMin((unsigned int*)addr, __float_as_uint(val));
}

// gfx1250 async global->LDS copy, 16 bytes per lane, tracked by ASYNCcnt.
__device__ __forceinline__ void async_b128(void* lds_ptr, const float* gptr) {
  unsigned off = (unsigned)(size_t)lds_ptr;   // low 32 bits of shared addr = LDS offset
  asm volatile("global_load_async_to_lds_b128 %0, %1, off"
               :: "v"(off), "v"(gptr) : "memory");
}

__device__ __forceinline__ void wait_async0() {
  asm volatile("s_wait_asynccnt 0x0" ::: "memory");
}

// =====================================================================
// Generic GEMM: C[M,N] = A[M,K] @ B[K,N] + bias  (fp32 in/out, bf16 WMMA)
// M%128==0, N%64==0, K%64==0. Block = 256 threads (8 waves), 128x64 tile.
// Wave w owns M-subtile w; 4 accumulators cover the 64 N columns.
// A is streamed via GLOBAL_LOAD_ASYNC_TO_LDS_B128 double-staged vs compute.
// =====================================================================
constexpr int BM = 128, BN = 64, BK = 64;

__global__ __launch_bounds__(256) void gemm_bias_wmma(
    const float* __restrict__ A, const float* __restrict__ Bm,
    const float* __restrict__ bias, float* __restrict__ C,
    int M, int N, int K)
{
  __shared__ float fA[BM * BK];   // raw fp32 A staging (async target)  32KB
  __shared__ bf16  bA[BM * BK];   // bf16 A [m][k], row stride 64       16KB
  __shared__ bf16  bB[BN * BK];   // bf16 B^T [n][k], row stride 64      8KB

  const int tilesN = N / BN;
  const int bm = blockIdx.x / tilesN;
  const int bn = blockIdx.x % tilesN;
  const int tid  = threadIdx.x;
  const int lane = tid & 31;
  const int wave = tid >> 5;        // 0..7 = M subtile
  const int laneHalf = lane >> 4;   // 0: lanes 0-15, 1: lanes 16-31
  const int lm = lane & 15;

  v8f acc[4] = {};

  // prologue: async-stage first A tile (8 x b128 per thread = 128x64 fp32)
#pragma unroll
  for (int i = 0; i < 8; ++i) {
    int w4 = (i * 256 + tid) * 4;
    int r = w4 >> 6, c = w4 & 63;
    async_b128(&fA[w4], &A[(size_t)(bm * BM + r) * K + c]);
  }

  for (int k0 = 0; k0 < K; k0 += BK) {
    // B tile: global fp32 (vector b128 loads) -> bf16 LDS transposed [n][k]
    // (overlaps with in-flight async A copy)
#pragma unroll
    for (int p = 0; p < 4; ++p) {
      int idx = p * 1024 + tid * 4;          // linear fp32 index in 64x64 tile
      int k = idx >> 6, n = idx & 63;        // 4 consecutive n in row k
      f4 v = *(const f4*)&Bm[(size_t)(k0 + k) * N + (bn * BN + n)];
      bB[(n + 0) * BK + k] = (bf16)v.x;
      bB[(n + 1) * BK + k] = (bf16)v.y;
      bB[(n + 2) * BK + k] = (bf16)v.z;
      bB[(n + 3) * BK + k] = (bf16)v.w;
    }
    wait_async0();         // A tile resident in LDS
    __syncthreads();

    // convert staged A tile fp32 -> bf16 (same [m][k] layout)
    {
      int base = tid * 32;
#pragma unroll
      for (int i = 0; i < 32; ++i) bA[base + i] = (bf16)fA[base + i];
    }
    __syncthreads();

    // stage NEXT A tile asynchronously while this tile's WMMAs run
    if (k0 + BK < K) {
#pragma unroll
      for (int i = 0; i < 8; ++i) {
        int w4 = (i * 256 + tid) * 4;
        int r = w4 >> 6, c = w4 & 63;
        async_b128(&fA[w4], &A[(size_t)(bm * BM + r) * K + (k0 + BK + c)]);
      }
    }

    // 8 WMMAs: 2 K-halves x 4 N-subtiles; preload all fragments, then issue
    // the 4 WMMAs back-to-back (single dscnt wait covers the batch).
#pragma unroll
    for (int kk = 0; kk < BK; kk += 32) {
      FragAB a;
#pragma unroll
      for (int v = 0; v < 8; ++v) {
        // A 16-bit 16x32 layout: lanes 0-15 K base 0/16, lanes 16-31 K base 8/24
        int ka = kk + ((v >> 2) << 4) + laneHalf * 8 + ((v & 3) << 1);
        a.u[v] = *(const unsigned int*)&bA[(wave * 16 + lm) * BK + ka];
      }
      FragAB b[4];
#pragma unroll
      for (int ns = 0; ns < 4; ++ns) {
#pragma unroll
        for (int v = 0; v < 8; ++v) {
          // B 16-bit 32x16 layout: lanes 0-15 K base 0/16, lanes 16-31 K base 8/24
          int kb2 = kk + ((v >> 2) << 4) + laneHalf * 8 + ((v & 3) << 1);
          b[ns].u[v] = *(const unsigned int*)&bB[(ns * 16 + lm) * BK + kb2];
        }
      }
#pragma unroll
      for (int ns = 0; ns < 4; ++ns) {
        acc[ns] = __builtin_amdgcn_wmma_f32_16x16x32_bf16(
            false, a.v, false, b[ns].v, (short)0, acc[ns], false, false);
      }
    }
    __syncthreads();
  }

  // epilogue: C/D layout -> VGPR r: lanes 0-15 M=r, lanes 16-31 M=r+8; N=lane&15
#pragma unroll
  for (int ns = 0; ns < 4; ++ns) {
    int n = bn * BN + ns * 16 + lm;
    float bv = bias ? bias[n] : 0.0f;
#pragma unroll
    for (int r = 0; r < 8; ++r) {
      int mrow = bm * BM + wave * 16 + r + laneHalf * 8;
      C[(size_t)mrow * N + n] = acc[ns][r] + bv;
    }
  }
}

// =====================================================================
// Embedding gathers: pe = pos_emb[pos_ids], de = dep_emb[dep_ids]
// =====================================================================
__global__ __launch_bounds__(256) void embed_gather(
    const int* __restrict__ pos_ids, const int* __restrict__ dep_ids,
    const float* __restrict__ pos_emb, const float* __restrict__ dep_emb,
    float* __restrict__ pe, float* __restrict__ de)
{
  int idx = blockIdx.x * 256 + threadIdx.x;       // over NTOK*HID (exact)
  int row = idx >> 8;
  int c   = idx & 255;
  pe[idx] = pos_emb[pos_ids[row] * HID + c];
  de[idx] = dep_emb[dep_ids[row] * HID + c];
}

// =====================================================================
// fused = 2*seq + sigmoid(q*k1)*pe + sigmoid(q*k2)*de ; x = relu(LN(fused))
// One wave per row (HID=256 -> 8 channels/lane).
// =====================================================================
__global__ __launch_bounds__(256) void fusion_ln_relu(
    const float* __restrict__ seq, const float* __restrict__ pe,
    const float* __restrict__ de,  const float* __restrict__ q,
    const float* __restrict__ k1,  const float* __restrict__ k2,
    const float* __restrict__ g,   const float* __restrict__ b,
    float* __restrict__ xout)
{
  int row  = blockIdx.x * 8 + (threadIdx.x >> 5);
  int lane = threadIdx.x & 31;
  size_t base = (size_t)row * HID;

  float f[8];
  float s = 0.0f;
#pragma unroll
  for (int j = 0; j < 8; ++j) {
    int c = j * 32 + lane;
    float sv = seq[base + c];
    float qv = q[base + c];
    float g1 = 1.0f / (1.0f + __expf(-qv * k1[base + c]));
    float g2 = 1.0f / (1.0f + __expf(-qv * k2[base + c]));
    f[j] = 2.0f * sv + g1 * pe[base + c] + g2 * de[base + c];
    s += f[j];
  }
  float mu = waveReduceSum(s) * (1.0f / HID);
  float vs = 0.0f;
#pragma unroll
  for (int j = 0; j < 8; ++j) { float d = f[j] - mu; vs += d * d; }
  float inv = rsqrtf(waveReduceSum(vs) * (1.0f / HID) + LN_EPS);
#pragma unroll
  for (int j = 0; j < 8; ++j) {
    int c = j * 32 + lane;
    float y = (f[j] - mu) * inv * g[c] + b[c];
    xout[base + c] = fmaxf(y, 0.0f);
  }
}

// =====================================================================
// gcn_in[r] = LN(x[word_token_idx[r]])   (wave per row)
// =====================================================================
__global__ __launch_bounds__(256) void gather_ln(
    const float* __restrict__ x, const int* __restrict__ wti,
    const float* __restrict__ g, const float* __restrict__ b,
    float* __restrict__ out)
{
  int row  = blockIdx.x * 8 + (threadIdx.x >> 5);
  int lane = threadIdx.x & 31;
  size_t src = (size_t)wti[row] * HID;
  size_t dst = (size_t)row * HID;

  float f[8];
  float s = 0.0f;
#pragma unroll
  for (int j = 0; j < 8; ++j) { f[j] = x[src + j * 32 + lane]; s += f[j]; }
  float mu = waveReduceSum(s) * (1.0f / HID);
  float vs = 0.0f;
#pragma unroll
  for (int j = 0; j < 8; ++j) { float d = f[j] - mu; vs += d * d; }
  float inv = rsqrtf(waveReduceSum(vs) * (1.0f / HID) + LN_EPS);
#pragma unroll
  for (int j = 0; j < 8; ++j) {
    int c = j * 32 + lane;
    out[dst + c] = (f[j] - mu) * inv * g[c] + b[c];
  }
}

// =====================================================================
// GAT init: mmax=-inf, den=0, outacc=0
// =====================================================================
__global__ __launch_bounds__(256) void init_gat(
    float* __restrict__ mmax, float* __restrict__ den, float* __restrict__ outacc)
{
  int idx = blockIdx.x * 256 + threadIdx.x;      // over NW*NHEADS*HID (exact)
  if (idx < NW * NHEADS) { mmax[idx] = -1e30f; den[idx] = 0.0f; }
  outacc[idx] = 0.0f;
}

__device__ __forceinline__ void edge_src_dst(const int* ei, int e, int& src, int& dst) {
  if (e < NE) { src = ei[e]; dst = ei[NE + e]; }
  else        { src = dst = e - NE; }            // self loops
}

// =====================================================================
// logit[e,h] = att[h] . leaky_relu(xl[src,h] + xr[dst,h]); segment max
// One wave per (edge, head).
// =====================================================================
__global__ __launch_bounds__(256) void edge_logit_kernel(
    const int* __restrict__ ei, const float* __restrict__ xl,
    const float* __restrict__ xr, const float* __restrict__ att,
    float* __restrict__ logit, float* __restrict__ mmax)
{
  int gid  = blockIdx.x * 8 + (threadIdx.x >> 5);  // over ETOT*NHEADS (exact)
  int lane = threadIdx.x & 31;
  int e = gid >> 2, h = gid & 3;
  int src, dst; edge_src_dst(ei, e, src, dst);
  size_t bl = ((size_t)src * NHEADS + h) * HID;
  size_t br = ((size_t)dst * NHEADS + h) * HID;
  float s = 0.0f;
#pragma unroll
  for (int j = 0; j < 8; ++j) {
    int c = j * 32 + lane;
    float v = xl[bl + c] + xr[br + c];
    v = (v > 0.0f) ? v : NEG_SLOPE * v;
    s += att[h * HID + c] * v;
  }
  s = waveReduceSum(s);
  if (lane == 0) {
    logit[gid] = s;
    atomicMaxFloat(&mmax[dst * NHEADS + h], s);
  }
}

// =====================================================================
// ex = exp(logit - max[dst]); den[dst] += ex   (thread per (e,h))
// =====================================================================
__global__ __launch_bounds__(256) void edge_exp_kernel(
    const int* __restrict__ ei, const float* __restrict__ logit,
    const float* __restrict__ mmax, float* __restrict__ exv,
    float* __restrict__ den)
{
  int gid = blockIdx.x * 256 + threadIdx.x;       // over ETOT*NHEADS (exact)
  int e = gid >> 2, h = gid & 3;
  int src, dst; edge_src_dst(ei, e, src, dst);
  (void)src;
  float ex = __expf(logit[gid] - mmax[dst * NHEADS + h]);
  exv[gid] = ex;
  atomicAdd(&den[dst * NHEADS + h], ex);
}

// =====================================================================
// outacc[dst,h] += (ex/den[dst,h]) * xl[src,h]   (wave per (e,h))
// =====================================================================
__global__ __launch_bounds__(256) void edge_accum_kernel(
    const int* __restrict__ ei, const float* __restrict__ exv,
    const float* __restrict__ den, const float* __restrict__ xl,
    float* __restrict__ outacc)
{
  int gid  = blockIdx.x * 8 + (threadIdx.x >> 5);  // over ETOT*NHEADS (exact)
  int lane = threadIdx.x & 31;
  int e = gid >> 2, h = gid & 3;
  int src, dst; edge_src_dst(ei, e, src, dst);
  float alpha = exv[gid] / den[dst * NHEADS + h];
  size_t bs = ((size_t)src * NHEADS + h) * HID;
  size_t bd = ((size_t)dst * NHEADS + h) * HID;
#pragma unroll
  for (int j = 0; j < 8; ++j) {
    int c = j * 32 + lane;
    atomicAdd(&outacc[bd + c], alpha * xl[bs + c]);
  }
}

// =====================================================================
// node epilogue: mean over heads + residual + bias -> relu -> LN -> Wc,bc
// One wave per node; writes [NW, 2] logits.
// =====================================================================
__global__ __launch_bounds__(256) void gat_final_kernel(
    const float* __restrict__ outacc, const float* __restrict__ resb,
    const float* __restrict__ gat_b, const float* __restrict__ g,
    const float* __restrict__ b, const float* __restrict__ Wc,
    const float* __restrict__ bc, float* __restrict__ out)
{
  int n    = blockIdx.x * 8 + (threadIdx.x >> 5);
  int lane = threadIdx.x & 31;
  size_t base = (size_t)n * NHEADS * HID;
  size_t rb   = (size_t)n * HID;

  float f[8];
  float s = 0.0f;
#pragma unroll
  for (int j = 0; j < 8; ++j) {
    int c = j * 32 + lane;
    float acc = 0.25f * (outacc[base + 0 * HID + c] + outacc[base + 1 * HID + c] +
                         outacc[base + 2 * HID + c] + outacc[base + 3 * HID + c]);
    acc += resb[rb + c] + gat_b[c];
    f[j] = fmaxf(acc, 0.0f);                      // relu BEFORE layer norm
    s += f[j];
  }
  float mu = waveReduceSum(s) * (1.0f / HID);
  float vs = 0.0f;
#pragma unroll
  for (int j = 0; j < 8; ++j) { float d = f[j] - mu; vs += d * d; }
  float inv = rsqrtf(waveReduceSum(vs) * (1.0f / HID) + LN_EPS);

  float p0 = 0.0f, p1 = 0.0f;
#pragma unroll
  for (int j = 0; j < 8; ++j) {
    int c = j * 32 + lane;
    float y = (f[j] - mu) * inv * g[c] + b[c];
    p0 += y * Wc[c * NLAB + 0];
    p1 += y * Wc[c * NLAB + 1];
  }
  p0 = waveReduceSum(p0);
  p1 = waveReduceSum(p1);
  if (lane == 0) {
    out[(size_t)n * NLAB + 0] = p0 + bc[0];
    out[(size_t)n * NLAB + 1] = p1 + bc[1];
  }
}

// =====================================================================
extern "C" void kernel_launch(void* const* d_in, const int* in_sizes, int n_in,
                              void* d_out, int out_size, void* d_ws, size_t ws_size,
                              hipStream_t stream)
{
  (void)in_sizes; (void)n_in; (void)out_size; (void)ws_size;

  const float* bert_out = (const float*)d_in[0];
  const int*   pos_ids  = (const int*)d_in[1];
  const int*   dep_ids  = (const int*)d_in[2];
  const int*   wti      = (const int*)d_in[3];
  const int*   ei       = (const int*)d_in[4];
  const float* W_red = (const float*)d_in[5];   const float* b_red = (const float*)d_in[6];
  const float* Wq  = (const float*)d_in[7];     const float* bq  = (const float*)d_in[8];
  const float* Wk1 = (const float*)d_in[9];     const float* bk1 = (const float*)d_in[10];
  const float* Wk2 = (const float*)d_in[11];    const float* bk2 = (const float*)d_in[12];
  const float* pos_emb = (const float*)d_in[13];
  const float* dep_emb = (const float*)d_in[14];
  const float* g_pre = (const float*)d_in[15];  const float* b_pre = (const float*)d_in[16];
  const float* g_cat = (const float*)d_in[17];  const float* b_cat = (const float*)d_in[18];
  const float* g_gcn = (const float*)d_in[19];  const float* b_gcn = (const float*)d_in[20];
  const float* Wl  = (const float*)d_in[21];    const float* bl  = (const float*)d_in[22];
  const float* Wr  = (const float*)d_in[23];    const float* br  = (const float*)d_in[24];
  const float* att = (const float*)d_in[25];
  const float* Wres = (const float*)d_in[26];
  const float* gat_b = (const float*)d_in[27];
  const float* Wc = (const float*)d_in[28];     const float* bc = (const float*)d_in[29];

  // workspace layout (floats)
  float* ws = (float*)d_ws;
  const size_t SM = (size_t)NTOK * HID;
  float* seq    = ws;
  float* pe     = seq + SM;
  float* de     = pe  + SM;
  float* q      = de  + SM;
  float* k1     = q   + SM;
  float* k2     = k1  + SM;
  float* x      = k2  + SM;
  float* gcn_in = x   + SM;
  float* xl     = gcn_in + (size_t)NW * HID;
  float* xr     = xl  + (size_t)NW * NHEADS * HID;
  float* resb   = xr  + (size_t)NW * NHEADS * HID;
  float* outacc = resb + (size_t)NW * HID;
  float* logit  = outacc + (size_t)NW * NHEADS * HID;
  float* exv    = logit + (size_t)ETOT * NHEADS;
  float* mmax   = exv + (size_t)ETOT * NHEADS;
  float* den    = mmax + (size_t)NW * NHEADS;

  dim3 blk(256);

  // tag embeddings
  embed_gather<<<NTOK * HID / 256, blk, 0, stream>>>(pos_ids, dep_ids, pos_emb, dep_emb, pe, de);

  // BERT reduction + projections (bf16 WMMA, f32 accumulate, async-staged A)
  gemm_bias_wmma<<<(NTOK / BM) * (HID / BN), blk, 0, stream>>>(bert_out, W_red, b_red, seq, NTOK, HID, DBERT);
  gemm_bias_wmma<<<(NTOK / BM) * (HID / BN), blk, 0, stream>>>(seq, Wq,  bq,  q,  NTOK, HID, HID);
  gemm_bias_wmma<<<(NTOK / BM) * (HID / BN), blk, 0, stream>>>(pe,  Wk1, bk1, k1, NTOK, HID, HID);
  gemm_bias_wmma<<<(NTOK / BM) * (HID / BN), blk, 0, stream>>>(de,  Wk2, bk2, k2, NTOK, HID, HID);

  // gated fusion + pre-LN + relu
  fusion_ln_relu<<<NTOK / 8, blk, 0, stream>>>(seq, pe, de, q, k1, k2, g_pre, b_pre, x);

  // subword->word gather + cat-LN
  gather_ln<<<NW / 8, blk, 0, stream>>>(x, wti, g_cat, b_cat, gcn_in);

  // GATv2 projections + residual
  gemm_bias_wmma<<<(NW / BM) * (NHEADS * HID / BN), blk, 0, stream>>>(gcn_in, Wl, bl, xl, NW, NHEADS * HID, HID);
  gemm_bias_wmma<<<(NW / BM) * (NHEADS * HID / BN), blk, 0, stream>>>(gcn_in, Wr, br, xr, NW, NHEADS * HID, HID);
  gemm_bias_wmma<<<(NW / BM) * (HID / BN),          blk, 0, stream>>>(gcn_in, Wres, nullptr, resb, NW, HID, HID);

  // edge softmax + aggregation
  init_gat<<<NW * NHEADS * HID / 256, blk, 0, stream>>>(mmax, den, outacc);
  edge_logit_kernel<<<ETOT * NHEADS / 8,  blk, 0, stream>>>(ei, xl, xr, att, logit, mmax);
  edge_exp_kernel  <<<ETOT * NHEADS / 256, blk, 0, stream>>>(ei, logit, mmax, exv, den);
  edge_accum_kernel<<<ETOT * NHEADS / 8,  blk, 0, stream>>>(ei, exv, den, xl, outacc);

  // head mean + residual + relu + gcn-LN + classifier
  gat_final_kernel<<<NW / 8, blk, 0, stream>>>(outacc, resb, gat_b, g_gcn, b_gcn, Wc, bc, (float*)d_out);
}
